// Petri_GraphConv_21947282882773
// MI455X (gfx1250) — matrix operand order repaired and checked by reading, not compiled, not run on hardware
//
#include <hip/hip_runtime.h>
#include <hip/hip_bf16.h>
#include <stdint.h>

// ---------------------------------------------------------------------------
// GraphConv GNN for MI455X (gfx1250, wave32).
//  - edge gather reads f16 activations via b64 vector loads (halves random
//    gather HBM traffic vs fp32)
//  - segment-sum via fp32 global atomics
//  - per-layer dual GEMM (agg@Wrel^T + x@Wroot^T):
//      * packed f16 weights staged into LDS by the Tensor Data Mover
//        (tensor_load_to_lds + s_wait_tensorcnt)
//      * one wave computes a full 16x128 output row-block: A fragments are
//        loaded once and reused across 8 register-blocked
//        v_wmma_f32_16x16x32_f16 accumulators (fp32 accumulate)
//  - readout MLP + scatter-mean in plain VALU (compute-trivial)
// ---------------------------------------------------------------------------

typedef _Float16     h16v __attribute__((ext_vector_type(16)));
typedef _Float16     h4v  __attribute__((ext_vector_type(4)));
typedef float        f8v  __attribute__((ext_vector_type(8)));
typedef unsigned int u32x4 __attribute__((ext_vector_type(4)));
typedef int          i32x8 __attribute__((ext_vector_type(8)));
typedef int          i32x4 __attribute__((ext_vector_type(4)));

#define HOUT   128
#define OTILES 8      // HOUT / 16

// ------------------------- elementwise helpers -----------------------------

__global__ void k_f32_to_f16(const float* __restrict__ in,
                             _Float16* __restrict__ out, size_t n) {
  size_t i = (size_t)blockIdx.x * blockDim.x + threadIdx.x;
  if (i < n) out[i] = (_Float16)in[i];
}

__global__ void k_zero_f32(float* __restrict__ p, size_t n) {
  size_t i = (size_t)blockIdx.x * blockDim.x + threadIdx.x;
  if (i < n) p[i] = 0.0f;
}

// ------------------------- weight packing (B layout) -----------------------
// Pack W[Hout,K] (row major, fp32) into a contiguous-k WMMA B operand order (same k order as the A loads) for
// V_WMMA_F32_16X16X32_F16:
//   lane l: N = l%16, Kbase = (l/16)*16; element j of the lane's 16 halves is
//   B[kchunk*32 + Kbase + j][N] = W[otile*16 + N][kchunk*32 + Kbase + j].
// Linear index: P[(((otile*kchunks + kc)*32 + lane)*16 + j]
__global__ void k_pack_b(const float* __restrict__ W,
                         _Float16* __restrict__ P, int K, int Hout) {
  int idx = blockIdx.x * blockDim.x + threadIdx.x;
  int total = Hout * K;
  if (idx >= total) return;
  int j    = idx & 15;
  int lane = (idx >> 4) & 31;
  int kch  = K >> 5;
  int kc   = (idx >> 9) % kch;
  int ot   = (idx >> 9) / kch;
  int n    = ot * 16 + (lane & 15);
  int k    = kc * 32 + (lane >> 4) * 16 + j;
  P[idx] = (_Float16)W[(size_t)n * K + k];
}

// ------------------------- edge scatter ------------------------------------
// agg[dst] += ew * xh[src]; one thread per (edge, 4 features).
__global__ void k_scatter(const int* __restrict__ src, const int* __restrict__ dst,
                          const float* __restrict__ ew,
                          const _Float16* __restrict__ xh,
                          float* __restrict__ agg, int E, int d) {
  int idx = blockIdx.x * blockDim.x + threadIdx.x;
  int perEdge = d >> 2;
  int e = idx / perEdge;
  if (e >= E) return;
  int f = (idx - e * perEdge) * 4;
  float w = ew[e];
  h4v xv = *(const h4v*)(xh + (size_t)src[e] * d + f);   // one b64 gather
  float* ap = agg + (size_t)dst[e] * d + f;
#pragma unroll
  for (int i = 0; i < 4; ++i) atomicAdd(ap + i, w * (float)xv[i]);
}

// ------------------------- WMMA dual-GEMM ----------------------------------
// out[M,HOUT] = Arel @ Brel + Aroot @ Broot + bias   (fp32 accumulate)
// Arel/Aroot: f16 [M,K] row major.
// Pw: packed weights per k_pack_b, [rel (HOUT*K) | root (HOUT*K)] halves,
//     staged to LDS by the Tensor Data Mover.
// One wave per 16-row block; register-blocked over all 8 otiles.
// M % 16 == 0, K % 32 == 0.
__global__ void k_gemm_wmma(const _Float16* __restrict__ Arel,
                            const _Float16* __restrict__ Aroot,
                            const _Float16* __restrict__ Pw,
                            const float* __restrict__ bias,
                            float* __restrict__ out,
                            int M, int K) {
  extern __shared__ _Float16 lds[];            // 2 * HOUT * K halves
  const int kchunks   = K >> 5;
  const int matHalves = HOUT * K;

  // ---- TDM: stage both packed weight matrices into LDS (wave 0 only) ----
  if ((threadIdx.x >> 5) == 0) {
    unsigned long long ga  = (unsigned long long)(uintptr_t)Pw;
    unsigned int ldsa      = (unsigned int)(uintptr_t)(void*)lds; // flat LDS offset bits
    unsigned int units8    = (unsigned int)((2u * (unsigned)matHalves * 2u) >> 3); // 8B units

    // D# group 0: count=1 (valid user descriptor), lds_addr, global_addr, type=2
    u32x4 g0;
    g0[0] = 1u;                                           // count=1, user mode
    g0[1] = ldsa;                                         // LDS byte address
    g0[2] = (unsigned int)(ga & 0xFFFFFFFFu);             // global_addr[31:0]
    g0[3] = (unsigned int)((ga >> 32) & 0x01FFFFFFu)      // global_addr[56:32]
          | (2u << 30);                                   // type = 2 ("image")
    // D# group 1: wg_mask=0, data_size=3 (8B), 1-D contiguous tile
    i32x8 g1 = {0, 0, 0, 0, 0, 0, 0, 0};
    g1[0] = (int)(3u << 16);                              // data_size = 8B
    g1[1] = (int)((units8 & 0xFFFFu) << 16);              // tensor_dim0 [15:0]
    g1[2] = (int)(units8 >> 16);                          // tensor_dim0 [31:16]
    g1[3] = (int)(units8 << 16);                          // tile_dim0 (<= 8192)
    // tile_dim1 = 0 (unused -> 1D), strides unused
    i32x4 g2 = {0, 0, 0, 0};
    i32x4 g3 = {0, 0, 0, 0};
    i32x8 g4 = {0, 0, 0, 0, 0, 0, 0, 0};                  // 6-arg form extra group
    __builtin_amdgcn_tensor_load_to_lds(g0, g1, g2, g3, g4, /*cpol=*/0);
    __builtin_amdgcn_s_wait_tensorcnt(0);
  }
  __syncthreads();

  int lane  = threadIdx.x & 31;
  int wv    = threadIdx.x >> 5;
  int mtile = blockIdx.x * (int)(blockDim.x >> 5) + wv;
  if (mtile * 16 >= M) return;                 // wave-uniform exit

  int mlocal  = lane & 15;
  int halfsel = lane >> 4;                     // K-half select per A layout
  size_t row = (size_t)(mtile * 16 + mlocal);

  f8v c[OTILES];
#pragma unroll
  for (int ot = 0; ot < OTILES; ++ot) c[ot] = (f8v){};

#pragma unroll
  for (int matsel = 0; matsel < 2; ++matsel) {
    const _Float16* A     = matsel ? Aroot : Arel;
    const _Float16* arow  = A + row * K;
    const _Float16* bbase = lds + (size_t)matsel * matHalves;
    for (int kc = 0; kc < kchunks; ++kc) {
      // A fragment: 16 contiguous halves, k = halfsel*16 + j (matches packed B).
      h16v a;
      const _Float16* ap0 = arow + kc * 32 + halfsel * 16;
      const _Float16* ap1 = ap0 + 8;
#pragma unroll
      for (int j = 0; j < 8; ++j) { a[j] = ap0[j]; a[8 + j] = ap1[j]; }
      // Reuse A across all 8 output tiles; B fragments from LDS.
#pragma unroll
      for (int ot = 0; ot < OTILES; ++ot) {
        h16v b;
        const _Float16* bp = bbase + ((((size_t)ot * kchunks + kc) * 32 + lane) << 4);
#pragma unroll
        for (int j = 0; j < 16; ++j) b[j] = bp[j];
        c[ot] = __builtin_amdgcn_wmma_f32_16x16x32_f16(
                    /*neg_a=*/false, a, /*neg_b=*/false, b,
                    /*c_mod=*/(short)0, c[ot],
                    /*reuse_a=*/false, /*reuse_b=*/false);
      }
    }
  }

  // C/D layout: VGPR r holds row (halfsel*8 + r), column = lane%16.
#pragma unroll
  for (int ot = 0; ot < OTILES; ++ot) {
    int col = ot * 16 + mlocal;
    float bv = bias[col];
#pragma unroll
    for (int r = 0; r < 8; ++r) {
      size_t orow = (size_t)(mtile * 16 + halfsel * 8 + r);
      out[orow * HOUT + col] = c[ot][r] + bv;
    }
  }
}

// ------------------------- readout MLP + scatter-mean ----------------------
__global__ void k_readout(const float* __restrict__ y,
                          const float* __restrict__ Wr1, const float* __restrict__ br1,
                          const float* __restrict__ Wr2, const float* __restrict__ br2,
                          const int* __restrict__ batch,
                          float* __restrict__ num, float* __restrict__ cnt, int N) {
  int n = blockIdx.x;
  if (n >= N) return;
  __shared__ float xs[128];
  __shared__ float hs[64];
  int t = threadIdx.x;
  xs[t] = y[(size_t)n * 128 + t];
  __syncthreads();
  if (t < 64) {
    float acc = br1[t];
    const float* wrow = Wr1 + (size_t)t * 128;
#pragma unroll 8
    for (int k = 0; k < 128; ++k) acc += wrow[k] * xs[k];
    hs[t] = fmaxf(acc, 0.0f);
  }
  __syncthreads();
  if (t == 0) {
    float acc = br2[0];
#pragma unroll 8
    for (int k = 0; k < 64; ++k) acc += Wr2[k] * hs[k];
    int g = batch[n];
    atomicAdd(&num[g], acc);
    atomicAdd(&cnt[g], 1.0f);
  }
}

__global__ void k_finalize(const float* __restrict__ num, const float* __restrict__ cnt,
                           float* __restrict__ out, int G) {
  int g = blockIdx.x * blockDim.x + threadIdx.x;
  if (g < G) out[g] = num[g] / fmaxf(cnt[g], 1.0f);
}

// ------------------------- host orchestration ------------------------------

static inline int cdiv_i(long long a, long long b) { return (int)((a + b - 1) / b); }

extern "C" void kernel_launch(void* const* d_in, const int* in_sizes, int n_in,
                              void* d_out, int out_size, void* d_ws, size_t ws_size,
                              hipStream_t stream) {
  const float* x        = (const float*)d_in[0];   // [N, IN]
  const int*   ei       = (const int*)  d_in[1];   // [2, E]
  const float* ew       = (const float*)d_in[2];   // [E]
  const int*   batch    = (const int*)  d_in[3];   // [N]
  const float* W_root0  = (const float*)d_in[4];   // [H, IN]
  const float* W_rel0   = (const float*)d_in[5];   // [H, IN]
  const float* b0       = (const float*)d_in[6];   // [H]
  const float* W_root_r = (const float*)d_in[7];   // [L, H, H]
  const float* W_rel_r  = (const float*)d_in[8];   // [L, H, H]
  const float* b_r      = (const float*)d_in[9];   // [L, H]
  const float* Wr1      = (const float*)d_in[10];  // [H/2, H]
  const float* br1      = (const float*)d_in[11];  // [H/2]
  const float* Wr2      = (const float*)d_in[12];  // [1, H/2]
  const float* br2      = (const float*)d_in[13];  // [1]

  const int N  = in_sizes[3];             // 100000
  const int E  = in_sizes[2];             // 1600000
  const int IN = in_sizes[0] / N;         // 64
  const int H  = 128;
  const int L  = in_sizes[9] / H;         // 3
  const int G  = out_size;                // 1024

  const int* srcp = ei;
  const int* dstp = ei + E;

  // --- workspace carve-out ---
  char* ws = (char*)d_ws;
  size_t off = 0;
  float*    bufA = (float*)(ws + off);    off += (size_t)N * H * sizeof(float);
  float*    bufB = (float*)(ws + off);    off += (size_t)N * H * sizeof(float);
  _Float16* xh   = (_Float16*)(ws + off); off += (size_t)N * H * sizeof(_Float16);
  _Float16* aggh = (_Float16*)(ws + off); off += (size_t)N * H * sizeof(_Float16);
  _Float16* Pw0  = (_Float16*)(ws + off); off += (size_t)2 * H * IN * sizeof(_Float16);
  _Float16* Pwr  = (_Float16*)(ws + off); off += (size_t)L * 2 * H * H * sizeof(_Float16);
  float*    num  = (float*)(ws + off);    off += (size_t)G * sizeof(float);
  float*    cnt  = (float*)(ws + off);    off += (size_t)G * sizeof(float);
  (void)ws_size; (void)n_in; (void)cnt;

  const int TB = 256;
  const int WAVES_PER_BLOCK = TB / 32;

  // --- pack weights into per-layer contiguous [rel | root] WMMA-B blocks ---
  k_pack_b<<<cdiv_i((long long)H * IN, TB), TB, 0, stream>>>(W_rel0,  Pw0,          IN, H);
  k_pack_b<<<cdiv_i((long long)H * IN, TB), TB, 0, stream>>>(W_root0, Pw0 + H * IN, IN, H);
  for (int l = 0; l < L; ++l) {
    _Float16* base = Pwr + (size_t)l * 2 * H * H;
    k_pack_b<<<cdiv_i((long long)H * H, TB), TB, 0, stream>>>(
        W_rel_r  + (size_t)l * H * H, base,         H, H);
    k_pack_b<<<cdiv_i((long long)H * H, TB), TB, 0, stream>>>(
        W_root_r + (size_t)l * H * H, base + H * H, H, H);
  }

  const int gemmBlocks = cdiv_i((long long)(N / 16), WAVES_PER_BLOCK);

  // --- layer 0: IN -> H ---
  {
    size_t nel = (size_t)N * IN;
    k_f32_to_f16<<<cdiv_i(nel, TB), TB, 0, stream>>>(x, xh, nel);
    k_zero_f32  <<<cdiv_i(nel, TB), TB, 0, stream>>>(bufB, nel);
    long long st = (long long)E * (IN / 4);
    k_scatter<<<cdiv_i(st, TB), TB, 0, stream>>>(srcp, dstp, ew, xh, bufB, E, IN);
    k_f32_to_f16<<<cdiv_i(nel, TB), TB, 0, stream>>>(bufB, aggh, nel);
    size_t ldsBytes = (size_t)2 * H * IN * sizeof(_Float16);   // 32 KB
    k_gemm_wmma<<<gemmBlocks, TB, ldsBytes, stream>>>(
        aggh, xh, Pw0, b0, bufA, N, IN);
  }

  // --- layers 1..L: H -> H ---
  for (int l = 0; l < L; ++l) {
    size_t nel = (size_t)N * H;
    k_f32_to_f16<<<cdiv_i(nel, TB), TB, 0, stream>>>(bufA, xh, nel);
    k_zero_f32  <<<cdiv_i(nel, TB), TB, 0, stream>>>(bufB, nel);
    long long st = (long long)E * (H / 4);
    k_scatter<<<cdiv_i(st, TB), TB, 0, stream>>>(srcp, dstp, ew, xh, bufB, E, H);
    k_f32_to_f16<<<cdiv_i(nel, TB), TB, 0, stream>>>(bufB, aggh, nel);
    size_t ldsBytes = (size_t)2 * H * H * sizeof(_Float16);    // 64 KB
    k_gemm_wmma<<<gemmBlocks, TB, ldsBytes, stream>>>(
        aggh, xh, Pwr + (size_t)l * 2 * H * H, b_r + (size_t)l * H, bufA, N, H);
  }

  // --- readout MLP + scatter-mean ---
  k_zero_f32<<<cdiv_i(2LL * G, TB), TB, 0, stream>>>(num, (size_t)2 * G); // num,cnt contiguous
  k_readout<<<N, 128, 0, stream>>>(bufA, Wr1, br1, Wr2, br2, batch, num, cnt, N);
  k_finalize<<<cdiv_i(G, TB), TB, 0, stream>>>(num, cnt, (float*)d_out, G);
}
